// PointNet2_43645457662639
// MI455X (gfx1250) — matrix operand order repaired and checked by reading, not compiled, not run on hardware
//
#include <hip/hip_runtime.h>
#include <math.h>

typedef __attribute__((ext_vector_type(16))) _Float16 v16h;
typedef __attribute__((ext_vector_type(8)))  _Float16 v8h;
typedef __attribute__((ext_vector_type(8)))  float    v8f;

#define EPSV 1e-5f
#define TWO_PI 6.283185307179586f

static inline int cdiv_h(int a, int b) { return (a + b - 1) / b; }
static inline int pad32_h(int x) { return (x + 31) & ~31; }

// ---------------------------------------------------------------------------
// WMMA tile primitives (gfx1250, wave32, v_wmma_f32_16x16x32_f16)
//
// A: activations, row-major [row][cinp] f16 (cinp multiple of 32).
//    This code's A lane layout (contiguous k, B packed to match): lane<16 holds row=lane, K in {0..15};
//    lane>=16 same row, K in {16..31}. Per K=32 chunk a lane needs two
//    contiguous 8-half runs: [kc*32 + kh*16 .. +7] and [kc*32+kh*16+8 .. +7].
//
// B: weights pre-packed so a lane's whole fragment is contiguous:
//    flat half index = ((kc*2 + kh)*cout + n)*16 + v*2 + j
//    i.e. k-pair kp = kc*16 + kh*8 + v, column n. Two v8h (b128) loads/lane.
//
// D: f32 accum (lane holds col n=lane%16, rows v + 8*(lane/16)).
// ---------------------------------------------------------------------------
__device__ __forceinline__ v16h load_a_frag(const _Float16* __restrict__ arow,
                                            int kc, int kh)
{
    const v8h a0 = *(const v8h*)(arow + kc * 32 + kh * 16);
    const v8h a1 = *(const v8h*)(arow + kc * 32 + kh * 16 + 8);
    return __builtin_shufflevector(a0, a1, 0, 1, 2, 3, 4, 5, 6, 7,
                                   8, 9, 10, 11, 12, 13, 14, 15);
}

__device__ __forceinline__ v16h load_b_frag(const _Float16* __restrict__ Wpk,
                                            int kc, int kh, int col, int cout)
{
    const _Float16* bp = Wpk + ((size_t)(kc * 2 + kh) * cout + col) * 16;
    const v8h b0 = *(const v8h*)(bp);
    const v8h b1 = *(const v8h*)(bp + 8);
    return __builtin_shufflevector(b0, b1, 0, 1, 2, 3, 4, 5, 6, 7,
                                   8, 9, 10, 11, 12, 13, 14, 15);
}

__device__ __forceinline__ void wmma_one_tile(
    const _Float16* __restrict__ in, _Float16* __restrict__ out,
    const _Float16* __restrict__ Wpk, const float* __restrict__ bias,
    int cinp, int cout, int mt, int nt, int lane)
{
    const int r   = lane & 15;
    const int kh  = lane >> 4;
    const int row = mt * 16 + r;
    const int col = nt * 16 + r;
    v8f acc = {0.f, 0.f, 0.f, 0.f, 0.f, 0.f, 0.f, 0.f};
    const int KC = cinp >> 5;
    const _Float16* arow = in + (size_t)row * cinp;
    for (int kc = 0; kc < KC; ++kc) {
        v16h av = load_a_frag(arow, kc, kh);
        v16h bv = load_b_frag(Wpk, kc, kh, col, cout);
        acc = __builtin_amdgcn_wmma_f32_16x16x32_f16(
            false, av, false, bv, (short)0, acc, false, false);
    }
    const float bb = bias[col];
#pragma unroll
    for (int u = 0; u < 8; ++u) {
        float y = acc[u] + bb;
        y = y > 0.f ? y : 0.f;
        out[(size_t)(mt * 16 + kh * 8 + u) * cout + col] = (_Float16)y;
    }
}

// Two M-tiles sharing one B fragment: halves weight traffic, gives two
// independent v_wmma ops per chunk for co-issue.
__device__ __forceinline__ void wmma_mtile_pair(
    const _Float16* __restrict__ in, _Float16* __restrict__ out,
    const _Float16* __restrict__ Wpk, const float* __restrict__ bias,
    int cinp, int cout, int mt0, int nt, int lane)
{
    const int r    = lane & 15;
    const int kh   = lane >> 4;
    const int row0 = mt0 * 16 + r;
    const int col  = nt * 16 + r;
    v8f acc0 = {0.f, 0.f, 0.f, 0.f, 0.f, 0.f, 0.f, 0.f};
    v8f acc1 = acc0;
    const int KC = cinp >> 5;
    const _Float16* arow0 = in + (size_t)row0 * cinp;
    const _Float16* arow1 = arow0 + (size_t)16 * cinp;
    for (int kc = 0; kc < KC; ++kc) {
        v16h bv  = load_b_frag(Wpk, kc, kh, col, cout);
        v16h av0 = load_a_frag(arow0, kc, kh);
        v16h av1 = load_a_frag(arow1, kc, kh);
        acc0 = __builtin_amdgcn_wmma_f32_16x16x32_f16(
            false, av0, false, bv, (short)0, acc0, false, false);
        acc1 = __builtin_amdgcn_wmma_f32_16x16x32_f16(
            false, av1, false, bv, (short)0, acc1, false, false);
    }
    const float bb = bias[col];
#pragma unroll
    for (int u = 0; u < 8; ++u) {
        float y0 = acc0[u] + bb;
        float y1 = acc1[u] + bb;
        y0 = y0 > 0.f ? y0 : 0.f;
        y1 = y1 > 0.f ? y1 : 0.f;
        out[(size_t)(mt0 * 16 + kh * 8 + u) * cout + col] = (_Float16)y0;
        out[(size_t)((mt0 + 1) * 16 + kh * 8 + u) * cout + col] = (_Float16)y1;
    }
}

__device__ __forceinline__ void wmma_layer_lds(
    const _Float16* in, _Float16* out,
    const _Float16* Wpk, const float* bias,
    int K, int cinp, int cout, int tid, int nth)
{
    const int w = tid >> 5, nw = nth >> 5, lane = tid & 31;
    const int MT = K >> 4, NT = cout >> 4;
    // wave-uniform tile loop => EXEC all-ones inside WMMA as required
    if ((MT & 1) == 0) {
        const int nt2 = (MT >> 1) * NT;
        for (int t = w; t < nt2; t += nw)
            wmma_mtile_pair(in, out, Wpk, bias, cinp, cout,
                            (t / NT) * 2, t % NT, lane);
    } else {
        for (int t = w; t < MT * NT; t += nw)
            wmma_one_tile(in, out, Wpk, bias, cinp, cout, t / NT, t % NT, lane);
    }
}

// ---------------------------------------------------------------------------
// BN-fold + f16 weight packing. W'[n][k] = W[n][k]*s, b' = s*(b-m)+bt,
// s = g*rsqrt(v+eps). Packed as ((kc*2+kh)*cout + n)*16 + v*2 + j halves,
// kp = kc*16 + kh*8 + v, zero-padded to cinp.
// ---------------------------------------------------------------------------
extern "C" __global__ void pn2_fold_layer(
    const float* __restrict__ W, const float* __restrict__ b,
    const float* __restrict__ g, const float* __restrict__ bt,
    const float* __restrict__ m, const float* __restrict__ v,
    _Float16* __restrict__ Wpk, float* __restrict__ bias,
    int cin, int cinp, int cout)
{
    int p = blockIdx.x * blockDim.x + threadIdx.x;   // half-pair index
    int KP = cinp >> 1;
    if (p >= cout * KP) return;
    int v8  = p & 7;
    int n   = (p >> 3) % cout;
    int t   = (p >> 3) / cout;
    int kh  = t & 1;
    int kc  = t >> 1;
    int kp  = kc * 16 + kh * 8 + v8;
    float s = g[n] * rsqrtf(v[n] + EPSV);
    int k0  = kp * 2;
    float w0 = (k0     < cin) ? W[(size_t)n * cin + k0]     * s : 0.f;
    float w1 = (k0 + 1 < cin) ? W[(size_t)n * cin + k0 + 1] * s : 0.f;
    Wpk[(size_t)p * 2]     = (_Float16)w0;
    Wpk[(size_t)p * 2 + 1] = (_Float16)w1;
    if (t == 0 && v8 == 0) bias[n] = s * (b[n] - m[n]) + bt[n];
}

// ---------------------------------------------------------------------------
// Split input [8,6,2048] -> coords [B,N,3], feats [B,N,3]
// ---------------------------------------------------------------------------
extern "C" __global__ void pn2_split(const float* __restrict__ x,
                                     float* __restrict__ coords,
                                     float* __restrict__ feats, int N)
{
    int i = blockIdx.x * blockDim.x + threadIdx.x;
    if (i >= 8 * N * 6) return;
    int b = i / (N * 6);
    int r = i % (N * 6);
    int n = r / 6, c = r % 6;
    float val = x[((size_t)b * 6 + c) * N + n];
    if (c < 3) coords[((size_t)b * N + n) * 3 + c] = val;
    else       feats[((size_t)b * N + n) * 3 + (c - 3)] = val;
}

// ---------------------------------------------------------------------------
// Farthest point sampling; one block per batch, dist lives in LDS
// ---------------------------------------------------------------------------
extern "C" __global__ void pn2_fps(const float* __restrict__ xyz,
                                   int* __restrict__ fidx, int N, int npoint)
{
    __shared__ float dist[2048];
    __shared__ float rd[256];
    __shared__ int   ri[256];
    const int b = blockIdx.x, tid = threadIdx.x, nth = blockDim.x;
    const float* p = xyz + (size_t)b * N * 3;
    for (int n = tid; n < N; n += nth) dist[n] = 1e10f;
    __syncthreads();
    int far = 0;
    for (int it = 0; it < npoint; ++it) {
        if (tid == 0) fidx[b * npoint + it] = far;
        float cx = p[far * 3 + 0], cy = p[far * 3 + 1], cz = p[far * 3 + 2];
        float best = -1.f; int bi = 0;
        for (int n = tid; n < N; n += nth) {
            float dx = p[n * 3] - cx, dy = p[n * 3 + 1] - cy, dz = p[n * 3 + 2] - cz;
            float d = dx * dx + dy * dy + dz * dz;
            float dd = dist[n];
            d = d < dd ? d : dd;
            dist[n] = d;
            if (d > best) { best = d; bi = n; }
        }
        rd[tid] = best; ri[tid] = bi;
        __syncthreads();
        for (int s = nth >> 1; s > 0; s >>= 1) {
            if (tid < s) {
                if (rd[tid + s] > rd[tid] ||
                    (rd[tid + s] == rd[tid] && ri[tid + s] < ri[tid])) {
                    rd[tid] = rd[tid + s]; ri[tid] = ri[tid + s];
                }
            }
            __syncthreads();
        }
        far = ri[0];
        __syncthreads();
    }
}

extern "C" __global__ void pn2_gather_xyz(const float* __restrict__ xyz,
                                          const int* __restrict__ fidx,
                                          float* __restrict__ newxyz,
                                          int N, int M, int total)
{
    int i = blockIdx.x * blockDim.x + threadIdx.x;
    if (i >= total) return;
    int b = i / (M * 3);
    int r = i % (M * 3);
    int m = r / 3, d = r % 3;
    newxyz[i] = xyz[((size_t)b * N + fidx[b * M + m]) * 3 + d];
}

// ---------------------------------------------------------------------------
// Ball query: one wave32 per centroid; ordered selection of first-K indices
// within radius (matches sort-then-take-K of the reference), fill with first.
// ---------------------------------------------------------------------------
extern "C" __global__ void pn2_ball_query(const float* __restrict__ xyz,
                                          const float* __restrict__ newxyz,
                                          int* __restrict__ idxout,
                                          int N, int M, int K, float r2)
{
    const int g = blockIdx.x;
    const int lane = threadIdx.x;
    const int b = g / M;
    const float qx = newxyz[g * 3], qy = newxyz[g * 3 + 1], qz = newxyz[g * 3 + 2];
    const float* p = xyz + (size_t)b * N * 3;
    int cnt = 0, first = 0;
    for (int base = 0; base < N && cnt < K; base += 32) {
        int n = base + lane;
        float dx = p[n * 3] - qx, dy = p[n * 3 + 1] - qy, dz = p[n * 3 + 2] - qz;
        bool inside = (dx * dx + dy * dy + dz * dz) <= r2;
#if __has_builtin(__builtin_amdgcn_ballot_w32)
        unsigned mask = __builtin_amdgcn_ballot_w32(inside);
#else
        unsigned mask = (unsigned)__ballot(inside);
#endif
        if (cnt == 0 && mask) first = base + __builtin_ctz(mask);
        int take = __builtin_popcount(mask);
        if (take > K - cnt) take = K - cnt;
        if (lane == 0) {
            unsigned mm = mask;
            for (int i2 = 0; i2 < take; ++i2) {
                int bit = __builtin_ctz(mm);
                mm &= mm - 1;
                idxout[(size_t)g * K + cnt + i2] = base + bit;
            }
        }
        cnt += take;
    }
    for (int t = cnt + lane; t < K; t += 32)
        idxout[(size_t)g * K + t] = first;
}

// ---------------------------------------------------------------------------
// Fused grouped 3-layer MLP + max-pool. One block per (batch, centroid).
// LDS ping-pong: bufA = K*max(cinp0, cout1) halves, bufB = K*max(cout0, cout2).
// ---------------------------------------------------------------------------
struct Mlp3Arg {
    const _Float16* W0; const float* b0; int cinp0; int cout0;
    const _Float16* W1; const float* b1; int cinp1; int cout1;
    const _Float16* W2; const float* b2; int cinp2; int cout2;
};

extern "C" __global__ void pn2_group_mlp(
    const float* __restrict__ xyz, const float* __restrict__ newxyz,
    const float* __restrict__ points, const int* __restrict__ idx,
    Mlp3Arg mlp, int N, int M, int K, int Cpts,
    float* __restrict__ out, int Ctot, int coff)
{
    extern __shared__ _Float16 smem[];
    const int g = blockIdx.x;
    const int b = g / M;
    const int tid = threadIdx.x, nth = blockDim.x;
    const int cin0 = Cpts + 3;
    const int cinp0 = mlp.cinp0;
    const int bufA_h = K * (cinp0 > mlp.cinp2 ? cinp0 : mlp.cinp2);
    _Float16* bufA = smem;
    _Float16* bufB = smem + bufA_h;

    // gather: [points(idx), xyz(idx)-new_xyz], zero-pad to cinp0, as f16
    for (int t = tid; t < K * cinp0; t += nth) {
        int j = t / cinp0, c = t % cinp0;
        int n = idx[(size_t)g * K + j];
        float val;
        if (c < Cpts) {
            val = points[((size_t)b * N + n) * Cpts + c];
        } else if (c < cin0) {
            int d = c - Cpts;
            val = xyz[((size_t)b * N + n) * 3 + d] - newxyz[(size_t)g * 3 + d];
        } else {
            val = 0.f;
        }
        bufA[(size_t)j * cinp0 + c] = (_Float16)val;
    }
    __syncthreads();

    __builtin_prefetch(mlp.W0, 0, 3);
    wmma_layer_lds(bufA, bufB, mlp.W0, mlp.b0, K, mlp.cinp0, mlp.cout0, tid, nth);
    __syncthreads();
    __builtin_prefetch(mlp.W1, 0, 3);
    wmma_layer_lds(bufB, bufA, mlp.W1, mlp.b1, K, mlp.cinp1, mlp.cout1, tid, nth);
    __syncthreads();
    __builtin_prefetch(mlp.W2, 0, 3);
    wmma_layer_lds(bufA, bufB, mlp.W2, mlp.b2, K, mlp.cinp2, mlp.cout2, tid, nth);
    __syncthreads();

    // max over K neighbors
    const int cf = mlp.cout2;
    for (int c = tid; c < cf; c += nth) {
        float mx = -3.4e38f;
        for (int j = 0; j < K; ++j) {
            float x = (float)bufB[(size_t)j * cf + c];
            mx = x > mx ? x : mx;
        }
        out[(size_t)g * Ctot + coff + c] = mx;
    }
}

// ---------------------------------------------------------------------------
// Global-memory WMMA GEMM + ReLU (sa3): one M-tile-pair per wave32 block
// ---------------------------------------------------------------------------
extern "C" __global__ void pn2_gemm_wmma(const _Float16* __restrict__ X,
                                         const _Float16* __restrict__ Wpk,
                                         const float* __restrict__ bias,
                                         _Float16* __restrict__ Y,
                                         int cinp, int cout)
{
    const int NT = cout >> 4;
    const int mt0 = (blockIdx.x / NT) * 2, nt = blockIdx.x % NT;
    __builtin_prefetch(Wpk, 0, 3);
    wmma_mtile_pair(X, Y, Wpk, bias, cinp, cout, mt0, nt, (int)(threadIdx.x & 31));
}

extern "C" __global__ void pn2_sa3_gather(const float* __restrict__ l2xyz,
                                          const float* __restrict__ l2pts,
                                          _Float16* __restrict__ X,
                                          int cinp, int Cpts, int total)
{
    int i = blockIdx.x * blockDim.x + threadIdx.x;
    if (i >= total) return;
    int row = i / cinp, c = i % cinp;
    float v;
    if (c < 3)             v = l2xyz[row * 3 + c];
    else if (c < 3 + Cpts) v = l2pts[(size_t)row * Cpts + (c - 3)];
    else                   v = 0.f;
    X[i] = (_Float16)v;
}

extern "C" __global__ void pn2_max_points(const _Float16* __restrict__ Y,
                                          float* __restrict__ l3,
                                          float* __restrict__ outl3,
                                          int P, int C, int total)
{
    int i = blockIdx.x * blockDim.x + threadIdx.x;
    if (i >= total) return;
    int b = i / C, c = i % C;
    float mx = -3.4e38f;
    for (int p = 0; p < P; ++p) {
        float x = (float)Y[((size_t)(b * P + p)) * C + c];
        mx = x > mx ? x : mx;
    }
    l3[i] = mx;
    outl3[i] = mx;   // l3[:, :, None] -> same flat order
}

// ---------------------------------------------------------------------------
// FC head (tiny, f32): dense+bn+relu, then tr / axis / state heads
// ---------------------------------------------------------------------------
extern "C" __global__ void pn2_dense_bn_relu(
    const float* __restrict__ X, const float* __restrict__ W,
    const float* __restrict__ b, const float* __restrict__ g,
    const float* __restrict__ bt, const float* __restrict__ m,
    const float* __restrict__ v, float* __restrict__ Y,
    int Bn, int cin, int cout)
{
    int i = blockIdx.x * blockDim.x + threadIdx.x;
    if (i >= Bn * cout) return;
    int bb = i / cout, n = i % cout;
    float y = b[n];
    const float* x = X + (size_t)bb * cin;
    const float* w = W + (size_t)n * cin;
    for (int k = 0; k < cin; ++k) y += x[k] * w[k];
    y = (y - m[n]) * rsqrtf(v[n] + EPSV) * g[n] + bt[n];
    Y[i] = y > 0.f ? y : 0.f;
}

extern "C" __global__ void pn2_head(const float* __restrict__ x2,
                                    const float* __restrict__ trW, const float* __restrict__ trB,
                                    const float* __restrict__ axW, const float* __restrict__ axB,
                                    const float* __restrict__ stW, const float* __restrict__ stB,
                                    float* __restrict__ out)
{
    const int B = 8, CIN = 256;
    int i = blockIdx.x * blockDim.x + threadIdx.x;
    if (i >= B * 15) return;
    int b = i / 15, u = i % 15;
    const float* x = x2 + (size_t)b * CIN;
    if (u < 9) {                         // translations
        float y = trB[u];
        for (int k = 0; k < CIN; ++k) y += x[k] * trW[u * CIN + k];
        out[b * 9 + u] = y;
    } else if (u < 12) {                 // axes per joint
        int j = u - 9;
        float yt = axB[2 * j], yp = axB[2 * j + 1];
        for (int k = 0; k < CIN; ++k) {
            yt += x[k] * axW[(2 * j) * CIN + k];
            yp += x[k] * axW[(2 * j + 1) * CIN + k];
        }
        float th = (1.f / (1.f + expf(-yt))) * TWO_PI;
        float ph = (1.f / (1.f + expf(-yp))) * TWO_PI;
        float* ax = out + 72;
        ax[b * 9 + j * 3 + 0] = sinf(th) * cosf(ph);
        ax[b * 9 + j * 3 + 1] = sinf(th) * sinf(ph);
        ax[b * 9 + j * 3 + 2] = cosf(th);
    } else {                             // joint states
        int j = u - 12;
        float y = stB[j];
        for (int k = 0; k < CIN; ++k) y += x[k] * stW[j * CIN + k];
        out[144 + b * 3 + j] = 1.f / (1.f + expf(-y));
    }
}

// ---------------------------------------------------------------------------
// Host orchestration
// ---------------------------------------------------------------------------
struct HostLayer {
    const float *W, *b, *g, *bt, *m, *v;
    int cin, cout, cinp;
    _Float16* Wpk;
    float* bias;
};

extern "C" void kernel_launch(void* const* d_in, const int* in_sizes, int n_in,
                              void* d_out, int out_size, void* d_ws, size_t ws_size,
                              hipStream_t stream)
{
    (void)in_sizes; (void)n_in; (void)out_size; (void)ws_size;
    auto F = [&](int i) { return (const float*)d_in[i]; };

    // ---- workspace arena ----
    char* ws = (char*)d_ws;
    size_t off = 0;
    auto alloc = [&](size_t bytes) -> void* {
        void* p = ws + off;
        off += (bytes + 255) & ~(size_t)255;
        return p;
    };

    const int B = 8, N0 = 2048;
    float* coords = (float*)alloc((size_t)B * N0 * 3 * 4);
    float* feats  = (float*)alloc((size_t)B * N0 * 3 * 4);
    int*   fps1   = (int*)  alloc((size_t)B * 512 * 4);
    float* l1xyz  = (float*)alloc((size_t)B * 512 * 3 * 4);
    float* l1pts  = (float*)alloc((size_t)B * 512 * 320 * 4);
    int*   fps2   = (int*)  alloc((size_t)B * 128 * 4);
    float* l2xyz  = (float*)alloc((size_t)B * 128 * 3 * 4);
    float* l2pts  = (float*)alloc((size_t)B * 128 * 640 * 4);
    const int sa1_K[3] = {16, 32, 128};
    const int sa2_K[3] = {32, 64, 128};
    const float sa1_r[3] = {0.1f, 0.2f, 0.4f};
    const float sa2_r[3] = {0.2f, 0.4f, 0.8f};
    int* idx1[3]; for (int s = 0; s < 3; ++s) idx1[s] = (int*)alloc((size_t)B * 512 * sa1_K[s] * 4);
    int* idx2[3]; for (int s = 0; s < 3; ++s) idx2[s] = (int*)alloc((size_t)B * 128 * sa2_K[s] * 4);
    _Float16* X0 = (_Float16*)alloc((size_t)1024 * 672 * 2);
    _Float16* Y1 = (_Float16*)alloc((size_t)1024 * 256 * 2);
    _Float16* Y2 = (_Float16*)alloc((size_t)1024 * 512 * 2);
    _Float16* Y3 = (_Float16*)alloc((size_t)1024 * 1024 * 2);
    float* l3 = (float*)alloc((size_t)B * 1024 * 4);
    float* x1 = (float*)alloc((size_t)B * 512 * 4);
    float* x2 = (float*)alloc((size_t)B * 256 * 4);

    // ---- layer configs (d_in order: xyz, sa1[3][3]*6, sa2[3][3]*6, sa3[3]*6, fc...) ----
    const int sa1_ch[3][3] = {{32, 32, 64}, {64, 64, 128}, {64, 96, 128}};
    const int sa2_ch[3][3] = {{64, 64, 128}, {128, 128, 256}, {128, 128, 256}};
    const int sa3_ch[3] = {256, 512, 1024};
    HostLayer L[21];
    int li = 0;
    for (int s = 0; s < 3; ++s) {
        int cin = 6;
        for (int l = 0; l < 3; ++l) {
            int i0 = 1 + (s * 3 + l) * 6;
            L[li] = {F(i0), F(i0 + 1), F(i0 + 2), F(i0 + 3), F(i0 + 4), F(i0 + 5),
                     cin, sa1_ch[s][l], pad32_h(cin), nullptr, nullptr};
            cin = sa1_ch[s][l]; ++li;
        }
    }
    for (int s = 0; s < 3; ++s) {
        int cin = 323;
        for (int l = 0; l < 3; ++l) {
            int i0 = 55 + (s * 3 + l) * 6;
            L[li] = {F(i0), F(i0 + 1), F(i0 + 2), F(i0 + 3), F(i0 + 4), F(i0 + 5),
                     cin, sa2_ch[s][l], pad32_h(cin), nullptr, nullptr};
            cin = sa2_ch[s][l]; ++li;
        }
    }
    {
        int cin = 643;
        for (int l = 0; l < 3; ++l) {
            int i0 = 109 + l * 6;
            L[li] = {F(i0), F(i0 + 1), F(i0 + 2), F(i0 + 3), F(i0 + 4), F(i0 + 5),
                     cin, sa3_ch[l], pad32_h(cin), nullptr, nullptr};
            cin = sa3_ch[l]; ++li;
        }
    }
    for (int i = 0; i < 21; ++i) {
        L[i].Wpk  = (_Float16*)alloc((size_t)L[i].cinp * L[i].cout * 2);
        L[i].bias = (float*)alloc((size_t)L[i].cout * 4);
        int tot = L[i].cout * (L[i].cinp >> 1);
        pn2_fold_layer<<<cdiv_h(tot, 256), 256, 0, stream>>>(
            L[i].W, L[i].b, L[i].g, L[i].bt, L[i].m, L[i].v,
            L[i].Wpk, L[i].bias, L[i].cin, L[i].cinp, L[i].cout);
    }

    auto mkarg = [&](const HostLayer* p) {
        Mlp3Arg a;
        a.W0 = p[0].Wpk; a.b0 = p[0].bias; a.cinp0 = p[0].cinp; a.cout0 = p[0].cout;
        a.W1 = p[1].Wpk; a.b1 = p[1].bias; a.cinp1 = p[1].cinp; a.cout1 = p[1].cout;
        a.W2 = p[2].Wpk; a.b2 = p[2].bias; a.cinp2 = p[2].cinp; a.cout2 = p[2].cout;
        return a;
    };
    auto group_smem = [&](int K, int cinp0, const int* ch) {
        int a = K * (cinp0 > ch[1] ? cinp0 : ch[1]);
        int bsz = K * (ch[0] > ch[2] ? ch[0] : ch[2]);
        return (size_t)(a + bsz) * sizeof(_Float16);
    };
    // allow >64KB dynamic LDS on the fused kernel (320KB/WGP on gfx1250)
    (void)hipFuncSetAttribute(reinterpret_cast<const void*>(pn2_group_mlp),
                              hipFuncAttributeMaxDynamicSharedMemorySize, 160 * 1024);

    // ---- stage 0: split input ----
    pn2_split<<<cdiv_h(B * N0 * 6, 256), 256, 0, stream>>>((const float*)d_in[0], coords, feats, N0);

    // ---- sa1 ----
    pn2_fps<<<B, 256, 0, stream>>>(coords, fps1, N0, 512);
    pn2_gather_xyz<<<cdiv_h(B * 512 * 3, 256), 256, 0, stream>>>(coords, fps1, l1xyz, N0, 512, B * 512 * 3);
    {
        int coff = 0;
        for (int s = 0; s < 3; ++s) {
            pn2_ball_query<<<B * 512, 32, 0, stream>>>(coords, l1xyz, idx1[s],
                                                       N0, 512, sa1_K[s], sa1_r[s] * sa1_r[s]);
            Mlp3Arg a = mkarg(&L[s * 3]);
            size_t sm = group_smem(sa1_K[s], L[s * 3].cinp, sa1_ch[s]);
            pn2_group_mlp<<<B * 512, 128, sm, stream>>>(coords, l1xyz, feats, idx1[s],
                                                        a, N0, 512, sa1_K[s], 3,
                                                        l1pts, 320, coff);
            coff += sa1_ch[s][2];
        }
    }

    // ---- sa2 ----
    pn2_fps<<<B, 256, 0, stream>>>(l1xyz, fps2, 512, 128);
    pn2_gather_xyz<<<cdiv_h(B * 128 * 3, 256), 256, 0, stream>>>(l1xyz, fps2, l2xyz, 512, 128, B * 128 * 3);
    {
        int coff = 0;
        for (int s = 0; s < 3; ++s) {
            pn2_ball_query<<<B * 128, 32, 0, stream>>>(l1xyz, l2xyz, idx2[s],
                                                       512, 128, sa2_K[s], sa2_r[s] * sa2_r[s]);
            Mlp3Arg a = mkarg(&L[9 + s * 3]);
            size_t sm = group_smem(sa2_K[s], L[9 + s * 3].cinp, sa2_ch[s]);
            pn2_group_mlp<<<B * 128, 128, sm, stream>>>(l1xyz, l2xyz, l1pts, idx2[s],
                                                        a, 512, 128, sa2_K[s], 320,
                                                        l2pts, 640, coff);
            coff += sa2_ch[s][2];
        }
    }

    // ---- sa3 (global WMMA GEMM chain, 1024 rows, M-tile pairs) ----
    pn2_sa3_gather<<<cdiv_h(1024 * 672, 256), 256, 0, stream>>>(l2xyz, l2pts, X0, 672, 640, 1024 * 672);
    pn2_gemm_wmma<<<(1024 / 32) * (256 / 16),  32, 0, stream>>>(X0, L[18].Wpk, L[18].bias, Y1, 672, 256);
    pn2_gemm_wmma<<<(1024 / 32) * (512 / 16),  32, 0, stream>>>(Y1, L[19].Wpk, L[19].bias, Y2, 256, 512);
    pn2_gemm_wmma<<<(1024 / 32) * (1024 / 16), 32, 0, stream>>>(Y2, L[20].Wpk, L[20].bias, Y3, 512, 1024);

    float* out = (float*)d_out;  // [trs(72) | axes(72) | states(24) | l3(8192)]
    pn2_max_points<<<cdiv_h(B * 1024, 256), 256, 0, stream>>>(Y3, l3, out + 168, 128, 1024, B * 1024);

    // ---- FC head ----
    pn2_dense_bn_relu<<<cdiv_h(B * 512, 128), 128, 0, stream>>>(
        l3, F(127), F(128), F(131), F(132), F(133), F(134), x1, B, 1024, 512);
    pn2_dense_bn_relu<<<cdiv_h(B * 256, 128), 128, 0, stream>>>(
        x1, F(129), F(130), F(135), F(136), F(137), F(138), x2, B, 512, 256);
    pn2_head<<<cdiv_h(B * 15, 64), 64, 0, stream>>>(
        x2, F(139), F(140), F(141), F(142), F(143), F(144), out);
}
